// TestMamba_32349693673990
// MI455X (gfx1250) — compile-verified, matrix operand order checked by reading, not run on hardware
//
#include <hip/hip_runtime.h>
#include <cstdint>

typedef __attribute__((ext_vector_type(16))) _Float16 v16h;
typedef __attribute__((ext_vector_type(8)))  _Float16 v8h;
typedef __attribute__((ext_vector_type(4)))  _Float16 h4;
typedef __attribute__((ext_vector_type(8)))  float    v8f;
typedef __attribute__((ext_vector_type(4)))  float    v4f;
typedef __attribute__((ext_vector_type(2)))  float    v2f;

#define NT        16384      // 8 * 2048 tokens
#define TSEQ      2048
#define BATCH     8
#define DMODEL    768
#define DINNER    1536
#define DSTATE    64
#define NHEADS    24
#define HEADDIM   64
#define CONVDIM   1664       // DINNER + 2*DSTATE
#define DINPROJ   3224       // 2*DINNER + 2*DSTATE + NHEADS
#define NCHUNK    32

__device__ __forceinline__ float siluf(float x) { return x / (1.f + __expf(-x)); }

#if __has_builtin(__builtin_amdgcn_wmma_f32_16x16x4_f32)
#define HAVE_WMMA_F32 1
__device__ __forceinline__ v8f wmma4(v2f a, v2f b, v8f c) {
  return __builtin_amdgcn_wmma_f32_16x16x4_f32(false, a, false, b, (short)0, c, false, false);
}
#else
#define HAVE_WMMA_F32 0
#endif

__device__ __forceinline__ v16h mk16(v8h lo, v8h hi) {
  return __builtin_shufflevector(lo, hi, 0,1,2,3,4,5,6,7,8,9,10,11,12,13,14,15);
}

// ---------------------------------------------------------------------------
// GEMM: C[M,N] = X[M,K] * W[N,K]^T (+ optional bias), f16 WMMA, f32 accum.
// Block tile 128x128, 256 threads = 8 wave32 waves; each wave owns
// 2 m-tiles x 4 n-tiles (8 accumulators, 8 wmma per K-step).
// K % 32 == 0, M % 128 == 0; N may be ragged.
// ---------------------------------------------------------------------------
__global__ __launch_bounds__(256) void gemm_f16_wmma(
    const float* __restrict__ X, const float* __restrict__ W,
    const float* __restrict__ bias, float* __restrict__ C,
    int M, int N, int K)
{
  __shared__ __align__(16) _Float16 As[128][40];   // pad 40 -> conflict-free b128
  __shared__ __align__(16) _Float16 Bs[128][40];

  const int tid  = threadIdx.x;
  const int lane = tid & 31;
  const int wid  = tid >> 5;
  const int l16  = lane & 15;
  const int hf   = lane >> 4;
  const int mt0  = (wid & 3) * 2;      // m-tiles mt0, mt0+1   (0..7)
  const int nt0  = (wid >> 2) * 4;     // n-tiles nt0..nt0+3   (0..7)
  const int m0   = blockIdx.x * 128;
  const int n0   = blockIdx.y * 128;
  const bool fullN = (n0 + 128 <= N);

  v8f acc[2][4];
#pragma unroll
  for (int mi = 0; mi < 2; ++mi)
#pragma unroll
    for (int ni = 0; ni < 4; ++ni)
      acc[mi][ni] = (v8f){0.f,0.f,0.f,0.f,0.f,0.f,0.f,0.f};

  // per-thread staging coords: 4 float4 chunks per matrix per K-step
  int srow[4], scol[4];
#pragma unroll
  for (int j = 0; j < 4; ++j) {
    int q = tid + 256 * j;             // 0..1023
    srow[j] = q >> 3;                  // 0..127
    scol[j] = (q & 7) << 2;            // 0,4,..,28
  }

  for (int k0 = 0; k0 < K; k0 += 32) {
    v4f xv[4], wv[4];
#pragma unroll
    for (int j = 0; j < 4; ++j)
      xv[j] = *(const v4f*)&X[(size_t)(m0 + srow[j]) * K + k0 + scol[j]];
    if (fullN) {
#pragma unroll
      for (int j = 0; j < 4; ++j)
        wv[j] = *(const v4f*)&W[(size_t)(n0 + srow[j]) * K + k0 + scol[j]];
    } else {
#pragma unroll
      for (int j = 0; j < 4; ++j) {
        int n = n0 + srow[j];
        v4f z = {0.f,0.f,0.f,0.f};
        wv[j] = (n < N) ? *(const v4f*)&W[(size_t)n * K + k0 + scol[j]] : z;
      }
    }
    if (k0 + 32 < K)                   // -> global_prefetch_b8
      __builtin_prefetch(&X[(size_t)(m0 + srow[0]) * K + (k0 + 32) + scol[0]], 0, 1);

    __syncthreads();                   // previous iteration's frag reads done
#pragma unroll
    for (int j = 0; j < 4; ++j) {
      *(h4*)&As[srow[j]][scol[j]] = __builtin_convertvector(xv[j], h4);
      *(h4*)&Bs[srow[j]][scol[j]] = __builtin_convertvector(wv[j], h4);
    }
    __syncthreads();

    v16h a[2], b[4];
#pragma unroll
    for (int mi = 0; mi < 2; ++mi) {
      const int mr = (mt0 + mi) * 16 + l16;
      a[mi] = mk16(*(const v8h*)&As[mr][hf * 8], *(const v8h*)&As[mr][hf * 8 + 16]);
    }
#pragma unroll
    for (int ni = 0; ni < 4; ++ni) {
      const int nr = (nt0 + ni) * 16 + l16;
      b[ni] = mk16(*(const v8h*)&Bs[nr][hf * 8], *(const v8h*)&Bs[nr][hf * 8 + 16]);
    }
#pragma unroll
    for (int mi = 0; mi < 2; ++mi)
#pragma unroll
      for (int ni = 0; ni < 4; ++ni)
        acc[mi][ni] = __builtin_amdgcn_wmma_f32_16x16x32_f16(
            false, a[mi], false, b[ni], (short)0, acc[mi][ni], false, false);
  }

#pragma unroll
  for (int mi = 0; mi < 2; ++mi) {
    const int mb = m0 + (mt0 + mi) * 16 + hf * 8;
#pragma unroll
    for (int ni = 0; ni < 4; ++ni) {
      const int nb = n0 + (nt0 + ni) * 16 + l16;
      if (nb < N) {
        const float bz = bias ? bias[nb] : 0.f;
#pragma unroll
        for (int r = 0; r < 8; ++r)
          C[(size_t)(mb + r) * N + nb] = acc[mi][ni][r] + bz;
      }
    }
  }
}

// ---------------------------------------------------------------------------
// SSD chunked scan, one block per (batch, head). 256 threads = 8 waves.
// All 64x64x64 matmuls on v_wmma_f32_16x16x4_f32 (f32-exact recurrence).
// ---------------------------------------------------------------------------
__global__ __launch_bounds__(256) void ssd_kernel(
    const float* __restrict__ xconv, const float* __restrict__ dtbuf,
    const float* __restrict__ A_log, const float* __restrict__ Dvec,
    float* __restrict__ yout)
{
  extern __shared__ float sm[];
  float* S   = sm;                 // [64][66] state, row p, col n
  float* Brs = S   + 64 * 66;      // [j][n]
  float* Crs = Brs + 64 * 66;      // [i][n]
  float* Dx  = Crs + 64 * 66;      // [j][p]
  float* Att = Dx  + 64 * 66;      // [i][j]
  float* dts = Att + 64 * 66;      // [64]
  float* seg = dts + 64;           // [64]
  float* wde = seg + 64;           // [64] exp(seg63 - seg_j)

  const int tid  = threadIdx.x;
  const int lane = tid & 31;
  const int wid  = tid >> 5;
  const int l16  = lane & 15;
  const int hf   = lane >> 4;
  const int rt   = wid & 3;           // wave's 16-row tile
  const int ct0  = (wid >> 2) * 2;    // wave's first 16-col tile

  const int h    = blockIdx.x % NHEADS;
  const int bidx = blockIdx.x / NHEADS;
  const float A  = -__expf(A_log[h]);
  const float Dp = Dvec[h];

  for (int e = tid; e < 64 * 64; e += 256) S[(e >> 6) * 66 + (e & 63)] = 0.f;
  __syncthreads();

  for (int c = 0; c < NCHUNK; ++c) {
    const int t0 = c * 64;
    if (tid < 64) dts[tid] = dtbuf[(size_t)(bidx * TSEQ + t0 + tid) * NHEADS + h];
    __syncthreads();
    if (tid < 64) {                                    // seg = cumsum(dt)*A
      float s = 0.f;
      for (int k2 = 0; k2 <= tid; ++k2) s += dts[k2];
      seg[tid] = s * A;
    }
    __syncthreads();
    if (tid < 64) wde[tid] = __expf(seg[63] - seg[tid]);
    for (int rr = 0; rr < 16; ++rr) {                  // load B, C, dt*x
      int e = tid + 256 * rr;
      int i = e >> 6, n = e & 63;
      const float* rowp = xconv + (size_t)(bidx * TSEQ + t0 + i) * CONVDIM;
      Brs[i * 66 + n] = rowp[DINNER + n];
      Crs[i * 66 + n] = rowp[DINNER + DSTATE + n];
      Dx [i * 66 + n] = dts[i] * rowp[h * HEADDIM + n];   // n plays p here
    }
    __syncthreads();

    // ---- scores = Cr . Br^T, then att = tril(exp(seg_i - seg_j)) * scores
    for (int u = 0; u < 2; ++u) {
      const int ti = rt, tj = ct0 + u;
      const int i = ti * 16 + l16;
      const int j = tj * 16 + l16;
      v8f acc = {0.f,0.f,0.f,0.f,0.f,0.f,0.f,0.f};
#if HAVE_WMMA_F32
      for (int ks = 0; ks < 16; ++ks) {
        int kq = ks * 4 + hf * 2;
        v2f a  = *(const v2f*)&Crs[i * 66 + kq];
        v2f bb = *(const v2f*)&Brs[j * 66 + kq];
        acc = wmma4(a, bb, acc);
      }
#else
      for (int r = 0; r < 8; ++r) {
        int ii = ti * 16 + hf * 8 + r;
        float s = 0.f;
        for (int k = 0; k < 64; ++k) s += Crs[ii * 66 + k] * Brs[j * 66 + k];
        acc[r] = s;
      }
#endif
      const int ib = ti * 16 + hf * 8;
      for (int r = 0; r < 8; ++r) {
        int ii = ib + r;
        Att[ii * 66 + j] = (j <= ii) ? __expf(seg[ii] - seg[j]) * acc[r] : 0.f;
      }
    }
    __syncthreads();

    // ---- y = att.dtx + (Cr.S^T)*exp(seg_i) + Dp*xs
    for (int u = 0; u < 2; ++u) {
      const int ti = rt, tp = ct0 + u;
      const int i = ti * 16 + l16;
      const int p = tp * 16 + l16;
      v8f ay = {0.f,0.f,0.f,0.f,0.f,0.f,0.f,0.f};
      v8f ai = {0.f,0.f,0.f,0.f,0.f,0.f,0.f,0.f};
#if HAVE_WMMA_F32
      for (int ks = 0; ks < 16; ++ks) {
        int kq = ks * 4 + hf * 2;
        v2f a = *(const v2f*)&Att[i * 66 + kq];
        v2f bb; bb[0] = Dx[kq * 66 + p]; bb[1] = Dx[(kq + 1) * 66 + p];
        ay = wmma4(a, bb, ay);
        a = *(const v2f*)&Crs[i * 66 + kq];
        v2f bs = *(const v2f*)&S[p * 66 + kq];
        ai = wmma4(a, bs, ai);
      }
#else
      for (int r = 0; r < 8; ++r) {
        int ii = ti * 16 + hf * 8 + r;
        float s1 = 0.f, s2 = 0.f;
        for (int k = 0; k < 64; ++k) {
          s1 += Att[ii * 66 + k] * Dx[k * 66 + p];
          s2 += Crs[ii * 66 + k] * S[p * 66 + k];
        }
        ay[r] = s1; ai[r] = s2;
      }
#endif
      const int ib = ti * 16 + hf * 8;
      for (int r = 0; r < 8; ++r) {
        int ii = ib + r;
        size_t tok = (size_t)(bidx * TSEQ) + t0 + ii;
        float xs = xconv[tok * CONVDIM + h * HEADDIM + p];
        float yv = ay[r] + ai[r] * __expf(seg[ii]) + Dp * xs;
        yout[tok * DINNER + h * HEADDIM + p] = yv;
      }
    }

    // ---- state update: S[p][n] = S*exp(seg63) + sum_j wde_j*dtx[j][p]*Br[j][n]
    const float cdec = __expf(seg[63]);
    v8f sacc[2];
    for (int u = 0; u < 2; ++u) {
      const int tp = rt, tn = ct0 + u;
      const int p  = tp * 16 + l16;
      const int n  = tn * 16 + l16;
      const int pb = tp * 16 + hf * 8;
      v8f acc;
      for (int r = 0; r < 8; ++r) acc[r] = S[(pb + r) * 66 + n] * cdec;
#if HAVE_WMMA_F32
      for (int ks = 0; ks < 16; ++ks) {
        int kq = ks * 4 + hf * 2;
        v2f a, bb;
        a[0]  = wde[kq]     * Dx[kq * 66 + p];
        a[1]  = wde[kq + 1] * Dx[(kq + 1) * 66 + p];
        bb[0] = Brs[kq * 66 + n];
        bb[1] = Brs[(kq + 1) * 66 + n];
        acc = wmma4(a, bb, acc);
      }
#else
      for (int r = 0; r < 8; ++r) {
        float s = 0.f;
        for (int k = 0; k < 64; ++k) s += wde[k] * Dx[k * 66 + pb + r] * Brs[k * 66 + n];
        acc[r] += s;
      }
#endif
      sacc[u] = acc;
    }
    __syncthreads();                 // all reads of S complete
    for (int u = 0; u < 2; ++u) {
      const int tp = rt, tn = ct0 + u;
      const int n  = tn * 16 + l16;
      const int pb = tp * 16 + hf * 8;
      for (int r = 0; r < 8; ++r) S[(pb + r) * 66 + n] = sacc[u][r];
    }
    __syncthreads();
  }
}

// ---------------------------------------------------------------------------
// Elementwise / small kernels
// ---------------------------------------------------------------------------
__global__ void dt_softplus_kernel(const float* __restrict__ zx,
                                   const float* __restrict__ dtb,
                                   float* __restrict__ dt)
{
  int idx = blockIdx.x * 256 + threadIdx.x;
  if (idx >= NT * NHEADS) return;
  int tok = idx / NHEADS, hh = idx % NHEADS;
  float v = zx[(size_t)tok * DINPROJ + (DINPROJ - NHEADS) + hh] + dtb[hh];
  dt[idx] = (v > 20.f) ? v : log1pf(__expf(v));
}

__global__ void convdw_silu_kernel(const float* __restrict__ zx,
                                   const float* __restrict__ cw,
                                   const float* __restrict__ cb,
                                   float* __restrict__ xc)
{
  int idx = blockIdx.x * 256 + threadIdx.x;        // over NT * (CONVDIM/4)
  if (idx >= NT * (CONVDIM / 4)) return;
  int tok = idx / (CONVDIM / 4);
  int c4  = (idx % (CONVDIM / 4)) * 4;
  int t   = tok & (TSEQ - 1);

  v4f acc = *(const v4f*)&cb[c4];
  v4f cwv[4];
#pragma unroll
  for (int u = 0; u < 4; ++u) cwv[u] = *(const v4f*)&cw[(c4 + u) * 4];
#pragma unroll
  for (int k = 0; k < 4; ++k) {
    int ts = t - 3 + k;
    if (ts >= 0) {
      v4f xv = *(const v4f*)&zx[(size_t)(tok + k - 3) * DINPROJ + DINNER + c4];
#pragma unroll
      for (int u = 0; u < 4; ++u) acc[u] += cwv[u][k] * xv[u];
    }
  }
  v4f o;
#pragma unroll
  for (int u = 0; u < 4; ++u) o[u] = siluf(acc[u]);
  *(v4f*)&xc[(size_t)tok * CONVDIM + c4] = o;
}

__global__ __launch_bounds__(256) void gate_rms_kernel(
    float* __restrict__ sy, const float* __restrict__ zx,
    const float* __restrict__ nw)
{
  __shared__ float red[256];
  const int tok = blockIdx.x, tid = threadIdx.x;
  size_t yb = (size_t)tok * DINNER, zb = (size_t)tok * DINPROJ;
  float yv[6]; float ss = 0.f;
#pragma unroll
  for (int j = 0; j < 6; ++j) {
    int cc = tid + 256 * j;
    float v = sy[yb + cc] * siluf(zx[zb + cc]);
    yv[j] = v; ss += v * v;
  }
  red[tid] = ss; __syncthreads();
  for (int s = 128; s > 0; s >>= 1) { if (tid < s) red[tid] += red[tid + s]; __syncthreads(); }
  float rms = rsqrtf(red[0] / (float)DINNER + 1e-5f);
#pragma unroll
  for (int j = 0; j < 6; ++j) {
    int cc = tid + 256 * j;
    sy[yb + cc] = yv[j] * rms * nw[cc];
  }
}

__global__ __launch_bounds__(256) void ln_kernel(
    const float* __restrict__ a, const float* __restrict__ res,
    float* __restrict__ out, const float* __restrict__ w,
    const float* __restrict__ bparam)
{
  __shared__ float red[256];
  const int tok = blockIdx.x, tid = threadIdx.x;
  size_t base = (size_t)tok * DMODEL;
  float v[3]; float s = 0.f;
#pragma unroll
  for (int j = 0; j < 3; ++j) {
    int cc = tid + 256 * j;
    float t = a[base + cc] + (res ? res[base + cc] : 0.f);
    v[j] = t; s += t;
  }
  red[tid] = s; __syncthreads();
  for (int st = 128; st > 0; st >>= 1) { if (tid < st) red[tid] += red[tid + st]; __syncthreads(); }
  float mean = red[0] / (float)DMODEL;
  __syncthreads();
  float ss = 0.f;
#pragma unroll
  for (int j = 0; j < 3; ++j) { float d = v[j] - mean; ss += d * d; }
  red[tid] = ss; __syncthreads();
  for (int st = 128; st > 0; st >>= 1) { if (tid < st) red[tid] += red[tid + st]; __syncthreads(); }
  float inv = rsqrtf(red[0] / (float)DMODEL + 1e-5f);
#pragma unroll
  for (int j = 0; j < 3; ++j) {
    int cc = tid + 256 * j;
    out[base + cc] = (v[j] - mean) * inv * w[cc] + bparam[cc];
  }
}

// ---------------------------------------------------------------------------
// Front-end convs
// ---------------------------------------------------------------------------
__global__ void front_pre_kernel(const float* __restrict__ x,
                                 const float* __restrict__ pw,
                                 const float* __restrict__ pb,
                                 float* __restrict__ h128)
{
  int idx = blockIdx.x * 256 + threadIdx.x;
  if (idx >= NT * 128) return;
  int bt = idx >> 7, o = idx & 127;
  float acc = pb[o];
  const float* xr = x + (size_t)bt * 64;
  const float* wr = pw + (size_t)o * 64;
#pragma unroll
  for (int i = 0; i < 64; i += 4) {
    v4f xv = *(const v4f*)&xr[i];
    v4f wv = *(const v4f*)&wr[i];
    acc += xv[0]*wv[0] + xv[1]*wv[1] + xv[2]*wv[2] + xv[3]*wv[3];
  }
  h128[idx] = siluf(acc);
}

__global__ __launch_bounds__(256) void front_conv_kernel(
    const float* __restrict__ h128,
    const float* __restrict__ w1, const float* __restrict__ b1,
    const float* __restrict__ w2, const float* __restrict__ b2,
    const float* __restrict__ w3, const float* __restrict__ b3,
    float* __restrict__ out)
{
  __shared__ float win[27][128];
  const int bt = blockIdx.x;
  const int b = bt >> 11, t = bt & (TSEQ - 1);
  const int tid = threadIdx.x;
  for (int e = tid; e < 27 * 128; e += 256) {
    int row = e >> 7, cc = e & 127;
    int ts = t + row - 13;
    win[row][cc] = (ts >= 0 && ts < TSEQ) ? h128[(size_t)((b << 11) + ts) * 128 + cc] : 0.f;
  }
  __syncthreads();
  for (int rr = 0; rr < 3; ++rr) {
    int o = tid + 256 * rr;
    const float* wsel; const float* bsel; int Kt, off, oo;
    if (o < 256)      { wsel = w1; bsel = b1; Kt = 3;  off = 12; oo = o; }
    else if (o < 512) { wsel = w2; bsel = b2; Kt = 9;  off = 9;  oo = o - 256; }
    else              { wsel = w3; bsel = b3; Kt = 27; off = 0;  oo = o - 512; }
    float acc = bsel[oo];
    for (int k = 0; k < Kt; ++k)
      for (int i2 = 0; i2 < 128; ++i2)
        acc += wsel[((size_t)oo * 128 + i2) * Kt + k] * win[off + k][i2];
    out[(size_t)bt * DMODEL + o] = siluf(acc);
  }
}

// ---------------------------------------------------------------------------
// Host orchestration
// ---------------------------------------------------------------------------
extern "C" void kernel_launch(void* const* d_in, const int* in_sizes, int n_in,
                              void* d_out, int out_size, void* d_ws, size_t ws_size,
                              hipStream_t stream)
{
  (void)in_sizes; (void)n_in; (void)out_size; (void)ws_size;
  const float* x       = (const float*)d_in[0];
  const float* pc_w    = (const float*)d_in[1];
  const float* pc_b    = (const float*)d_in[2];
  const float* c1_w    = (const float*)d_in[3];
  const float* c1_b    = (const float*)d_in[4];
  const float* c2_w    = (const float*)d_in[5];
  const float* c2_b    = (const float*)d_in[6];
  const float* c3_w    = (const float*)d_in[7];
  const float* c3_b    = (const float*)d_in[8];
  const float* in_proj = (const float*)d_in[9];
  const float* conv_w  = (const float*)d_in[10];
  const float* conv_b  = (const float*)d_in[11];
  const float* dt_bias = (const float*)d_in[12];
  const float* A_log   = (const float*)d_in[13];
  const float* Dvec    = (const float*)d_in[14];
  const float* norm_w  = (const float*)d_in[15];
  const float* out_prj = (const float*)d_in[16];
  const float* ln_w    = (const float*)d_in[17];
  const float* ln_b    = (const float*)d_in[18];
  const float* fn_w    = (const float*)d_in[19];
  const float* fn_b    = (const float*)d_in[20];
  const float* lo_w    = (const float*)d_in[21];
  const float* lo_b    = (const float*)d_in[22];
  float* out = (float*)d_out;

  // Workspace layout (floats): ~523 MB total
  float* ws    = (float*)d_ws;
  float* hbuf  = ws;                                  // NT*768  residual stream
  float* ybuf  = hbuf  + (size_t)NT * DMODEL;         // NT*768  layer output
  float* zx    = ybuf  + (size_t)NT * DMODEL;         // NT*3224 in_proj out
  float* xconv = zx    + (size_t)NT * DINPROJ;        // NT*1664 conv+silu
  float* dtbuf = xconv + (size_t)NT * CONVDIM;        // NT*24   softplus(dt)
  float* sy    = dtbuf + (size_t)NT * NHEADS;         // NT*1536 ssd out / gated
  float* h128  = sy;                                  // alias (front stage only)

  const size_t SSD_LDS = (size_t)(5 * 64 * 66 + 3 * 64) * sizeof(float);

  // Front-end
  front_pre_kernel<<<(NT * 128 + 255) / 256, 256, 0, stream>>>(x, pc_w, pc_b, h128);
  front_conv_kernel<<<NT, 256, 0, stream>>>(h128, c1_w, c1_b, c2_w, c2_b, c3_w, c3_b, hbuf);

  // Mamba-2 layers
  for (int l = 0; l < 4; ++l) {
    gemm_f16_wmma<<<dim3(NT / 128, (DINPROJ + 127) / 128), 256, 0, stream>>>(
        hbuf, in_proj + (size_t)l * DINPROJ * DMODEL, nullptr, zx, NT, DINPROJ, DMODEL);
    dt_softplus_kernel<<<(NT * NHEADS + 255) / 256, 256, 0, stream>>>(
        zx, dt_bias + l * NHEADS, dtbuf);
    convdw_silu_kernel<<<(NT * (CONVDIM / 4) + 255) / 256, 256, 0, stream>>>(
        zx, conv_w + (size_t)l * CONVDIM * 4, conv_b + (size_t)l * CONVDIM, xconv);
    ssd_kernel<<<BATCH * NHEADS, 256, SSD_LDS, stream>>>(
        xconv, dtbuf, A_log + l * NHEADS, Dvec + l * NHEADS, sy);
    gate_rms_kernel<<<NT, 256, 0, stream>>>(sy, zx, norm_w + (size_t)l * DINNER);
    gemm_f16_wmma<<<dim3(NT / 128, DMODEL / 128), 256, 0, stream>>>(
        sy, out_prj + (size_t)l * DMODEL * DINNER, nullptr, ybuf, NT, DMODEL, DINNER);
    ln_kernel<<<NT, 256, 0, stream>>>(hbuf, ybuf, hbuf, ln_w + l * DMODEL, ln_b + l * DMODEL);
  }

  // Final norm + classifier head
  ln_kernel<<<NT, 256, 0, stream>>>(hbuf, nullptr, ybuf, fn_w, fn_b);
  gemm_f16_wmma<<<dim3(NT / 128, 1), 256, 0, stream>>>(ybuf, lo_w, lo_b, out, NT, 10, DMODEL);
}
